// Base_Net_33500744909482
// MI455X (gfx1250) — matrix operand order repaired and checked by reading, not compiled, not run on hardware
//
#include <hip/hip_runtime.h>
#include <hip/hip_bf16.h>

typedef __attribute__((ext_vector_type(16))) _Float16 v16h;
typedef __attribute__((ext_vector_type(8)))  float    v8f;
typedef __attribute__((ext_vector_type(4)))  unsigned int v4u;

union Frag16 { v16h h; v4u u[2]; };

// ---------------- WMMA fragment loads (wave32, per CDNA5 ISA layouts) -------
// A (16x32 f16, row-major source, leading dim = ld):
//   lanes 0-15 : row M=lane,   K = kb+0..7  (v0-3), K = kb+16..23 (v4-7)
//   lanes16-31 : row M=lane-16,K = kb+8..15 (v0-3), K = kb+24..31 (v4-7)
__device__ inline v16h load_fragA(const _Float16* base, int ld, int row0,
                                  int kb, int lane) {
    int r  = row0 + (lane & 15);
    const _Float16* p = base + (size_t)r * ld + kb + ((lane >> 4) << 3);
    Frag16 f;
    f.u[0] = *(const v4u*)(p);        // 8 halves
    f.u[1] = *(const v4u*)(p + 16);   // 8 halves
    return f.h;
}
// B (32x16 f16). Source is pre-transposed weights BT[Ncol][K]; same K order as A:
//   lanes 0-15 : col N=lane,    K = kb+0..7  (v0-3), K = kb+16..23 (v4-7)
//   lanes16-31 : col N=lane-16, K = kb+8..15 (v0-3), K = kb+24..31 (v4-7)
__device__ inline v16h load_fragB(const _Float16* bt, int ld, int col0,
                                  int kb, int lane) {
    int c  = col0 + (lane & 15);
    const _Float16* p = bt + (size_t)c * ld + kb + ((lane >> 4) << 3);
    Frag16 f;
    f.u[0] = *(const v4u*)(p);
    f.u[1] = *(const v4u*)(p + 16);
    return f.h;
}

// ---------------- GEMM: C[M,Nc] = A[M,K] * BT[Nc,K]^T  (f16 in, f32 out) ----
// one wave -> 16(M) x 64(N) tile; A fragment reused across 4 WMMAs per k-step
__global__ __launch_bounds__(256)
void gemm_wmma(const _Float16* __restrict__ A, const _Float16* __restrict__ BT,
               float* __restrict__ C, int M, int Nc, int K) {
    int wave = (blockIdx.x * blockDim.x + threadIdx.x) >> 5;
    int lane = threadIdx.x & 31;
    int ntiles = Nc >> 6;
    int tm = (wave / ntiles) << 4;
    int tn = (wave % ntiles) << 6;
    if (tm >= M) return;                    // wave-uniform exit (EXEC stays all-1)
    v8f c0 = {}, c1 = {}, c2 = {}, c3 = {};
    for (int kb = 0; kb < K; kb += 32) {
        v16h a  = load_fragA(A,  K, tm,      kb, lane);
        v16h b0 = load_fragB(BT, K, tn,      kb, lane);
        v16h b1 = load_fragB(BT, K, tn + 16, kb, lane);
        v16h b2 = load_fragB(BT, K, tn + 32, kb, lane);
        v16h b3 = load_fragB(BT, K, tn + 48, kb, lane);
        c0 = __builtin_amdgcn_wmma_f32_16x16x32_f16(false, a, false, b0, (short)0, c0, false, false);
        c1 = __builtin_amdgcn_wmma_f32_16x16x32_f16(false, a, false, b1, (short)0, c1, false, false);
        c2 = __builtin_amdgcn_wmma_f32_16x16x32_f16(false, a, false, b2, (short)0, c2, false, false);
        c3 = __builtin_amdgcn_wmma_f32_16x16x32_f16(false, a, false, b3, (short)0, c3, false, false);
    }
    // D layout: VGPR r -> M=r (lanes 0-15), M=8+r (lanes 16-31); N = lane&15
    int rowoff = tm + ((lane >> 4) << 3);
    int col = lane & 15;
    #pragma unroll
    for (int r = 0; r < 8; ++r) {
        size_t base = (size_t)(rowoff + r) * Nc + tn + col;
        C[base]      = c0[r];
        C[base + 16] = c1[r];
        C[base + 32] = c2[r];
        C[base + 48] = c3[r];
    }
}

// ---------------- small helpers --------------------------------------------
__device__ inline unsigned int float_key(float f) {   // order-preserving uint
    unsigned int u = __float_as_uint(f);
    return (u & 0x80000000u) ? ~u : (u | 0x80000000u);
}
__device__ inline float key_float(unsigned int k) {
    unsigned int u = (k & 0x80000000u) ? (k & 0x7FFFFFFFu) : ~k;
    return __uint_as_float(u);
}

__global__ void f32_to_f16(const float* __restrict__ in, _Float16* __restrict__ out,
                           long long n) {
    long long t = (long long)blockIdx.x * blockDim.x + threadIdx.x;
    if (t < n) out[t] = (_Float16)in[t];
}

// W_heads [8][128][64] f32 -> WT1 [512][128] f16 : WT1[n][k] = W[n>>6][k][n&63]
__global__ void pack_W1(const float* __restrict__ W, _Float16* __restrict__ WT) {
    int t = blockIdx.x * blockDim.x + threadIdx.x;
    if (t >= 512 * 128) return;
    int n = t >> 7, k = t & 127;
    WT[t] = (_Float16)W[((n >> 6) * 128 + k) * 64 + (n & 63)];
}
// W_out [512][128] f32 -> WT2 [128][512] f16 : WT2[n][k] = W[k][n]
__global__ void pack_W2(const float* __restrict__ W, _Float16* __restrict__ WT) {
    int t = blockIdx.x * blockDim.x + threadIdx.x;
    if (t >= 128 * 512) return;
    int n = t >> 9, k = t & 511;
    WT[t] = (_Float16)W[k * 128 + n];
}

// s_src[n,h] = h_row . a_h[0:D], s_tgt[n,h] = h_row . a_h[D:2D]
__global__ void node_scores(const float* __restrict__ Hm, const float* __restrict__ a,
                            float* __restrict__ s_src, float* __restrict__ s_tgt,
                            int N, int H, int D, int rowD, int astride) {
    int t = blockIdx.x * blockDim.x + threadIdx.x;
    if (t >= N * H) return;
    int node = t / H, h = t - node * H;
    const float* row = Hm + (size_t)node * rowD + h * D;
    const float* ah  = a + h * astride;
    float s1 = 0.f, s2 = 0.f;
    for (int i = 0; i < D; ++i) { float v = row[i]; s1 += v * ah[i]; s2 += v * ah[D + i]; }
    s_src[t] = s1; s_tgt[t] = s2;
}

// score[e,h] = leaky_relu(s_src[src]+s_tgt[tgt]+ea.a_e) ; global per-head max
__global__ void edge_scores(const long long* __restrict__ ei, const float* __restrict__ eattr,
                            const float* __restrict__ a, const float* __restrict__ s_src,
                            const float* __restrict__ s_tgt, float* __restrict__ score,
                            unsigned int* __restrict__ maxkeys,
                            int E, int H, int D, int astride) {
    __shared__ float ae[8][16];
    __shared__ unsigned int smax[8];
    int tid = threadIdx.x;
    if (tid < H * 16) ae[tid >> 4][tid & 15] = a[(tid >> 4) * astride + 2 * D + (tid & 15)];
    if (tid < H) smax[tid] = 0u;
    __syncthreads();
    int e = blockIdx.x * blockDim.x + tid;
    if (e < E) {
        int s = (int)ei[e], g = (int)ei[E + e];
        float ea[16];
        const float* ep = eattr + (size_t)e * 16;
        #pragma unroll
        for (int i = 0; i < 16; ++i) ea[i] = ep[i];
        for (int h = 0; h < H; ++h) {
            float se = 0.f;
            #pragma unroll
            for (int i = 0; i < 16; ++i) se += ea[i] * ae[h][i];
            float raw = s_src[s * H + h] + s_tgt[g * H + h] + se;
            float sc  = raw > 0.f ? raw : 0.01f * raw;
            score[(size_t)e * H + h] = sc;
            atomicMax(&smax[h], float_key(sc));
        }
    }
    __syncthreads();
    if (tid < H) atomicMax(&maxkeys[tid], smax[tid]);
}

// ex = exp(score - max[h]); denom[src,h] += ex   (score overwritten with ex)
__global__ void exp_denom(float* __restrict__ score, const unsigned int* __restrict__ maxkeys,
                          const long long* __restrict__ ei, float* __restrict__ denom,
                          int E, int H) {
    int t = blockIdx.x * blockDim.x + threadIdx.x;
    if (t >= E * H) return;
    int e = t / H, h = t - e * H;
    float ex = expf(score[t] - key_float(maxkeys[h]));
    score[t] = ex;
    atomicAdd(&denom[(int)ei[e] * H + h], ex);
}

// hp[src, :] += h[tgt, :] * attn   (one float4 per thread)
__global__ void aggregate(const long long* __restrict__ ei, const float* __restrict__ Hm,
                          const float* __restrict__ ex, const float* __restrict__ denom,
                          float* __restrict__ hp, int E, int D, int Dh, int H) {
    long long t = (long long)blockIdx.x * blockDim.x + threadIdx.x;
    int q4 = D >> 2;
    if (t >= (long long)E * q4) return;
    int e = (int)(t / q4);
    int d = (int)(t - (long long)e * q4) << 2;
    int h = d / Dh;
    int s = (int)ei[e], g = (int)ei[E + e];
    float attn = ex[(size_t)e * H + h] / (denom[s * H + h] + 1e-16f);
    const float4 hv = *(const float4*)(Hm + (size_t)g * D + d);
    float* out = hp + (size_t)s * D + d;
    atomicAdd(out + 0, hv.x * attn);
    atomicAdd(out + 1, hv.y * attn);
    atomicAdd(out + 2, hv.z * attn);
    atomicAdd(out + 3, hv.w * attn);
}

// x = elu(elu(v)) -> f16 (feeds layer-2 WMMA GEMM)
__global__ void elu2_to_f16(const float* __restrict__ in, _Float16* __restrict__ out,
                            long long n) {
    long long t = (long long)blockIdx.x * blockDim.x + threadIdx.x;
    if (t >= n) return;
    float v = in[t];
    v = v > 0.f ? v : expm1f(v);
    v = v > 0.f ? v : expm1f(v);
    out[t] = (_Float16)v;
}

// out[n,:] = elu(hp[n,:]) - logsumexp(elu(hp[n,:])) ; one wave per node (D=128)
__global__ void elu_logsoftmax(const float* __restrict__ hp, float* __restrict__ out, int N) {
    int wave = (blockIdx.x * blockDim.x + threadIdx.x) >> 5;
    int lane = threadIdx.x & 31;
    if (wave >= N) return;
    const float* row = hp + (size_t)wave * 128;
    float v[4];
    #pragma unroll
    for (int j = 0; j < 4; ++j) {
        float x = row[lane * 4 + j];
        v[j] = x > 0.f ? x : expm1f(x);
    }
    float m = fmaxf(fmaxf(v[0], v[1]), fmaxf(v[2], v[3]));
    for (int o = 16; o >= 1; o >>= 1) m = fmaxf(m, __shfl_xor(m, o, 32));
    float sum = 0.f;
    #pragma unroll
    for (int j = 0; j < 4; ++j) sum += expf(v[j] - m);
    for (int o = 16; o >= 1; o >>= 1) sum += __shfl_xor(sum, o, 32);
    float lse = m + logf(sum);
    #pragma unroll
    for (int j = 0; j < 4; ++j) out[(size_t)wave * 128 + lane * 4 + j] = v[j] - lse;
}

// ---------------------------------------------------------------------------
extern "C" void kernel_launch(void* const* d_in, const int* in_sizes, int n_in,
                              void* d_out, int out_size, void* d_ws, size_t ws_size,
                              hipStream_t stream) {
    const float*      X       = (const float*)d_in[0];
    const long long*  ei      = (const long long*)d_in[1];   // [2,E] int64
    const float*      eattr   = (const float*)d_in[2];       // [E,16]
    const float*      W_heads = (const float*)d_in[3];       // [8,128,64]
    const float*      a_heads = (const float*)d_in[4];       // [8,144]
    const float*      W_out   = (const float*)d_in[5];       // [512,128]
    const float*      a_out   = (const float*)d_in[6];       // [272]
    float*            out     = (float*)d_out;

    const int N = in_sizes[0] / 128;
    const int E = in_sizes[1] / 2;

    // carve workspace (256B aligned slices)
    char* p = (char*)d_ws;
    auto carve = [&](size_t bytes) { char* r = p; p += (bytes + 255) & ~(size_t)255; return r; };
    _Float16* Xh  = (_Float16*)carve((size_t)N * 128 * 2);
    _Float16* WT1 = (_Float16*)carve((size_t)512 * 128 * 2);
    _Float16* WT2 = (_Float16*)carve((size_t)128 * 512 * 2);
    float* h1  = (float*)carve((size_t)N * 512 * 4);
    float* ss1 = (float*)carve((size_t)N * 8 * 4);
    float* st1 = (float*)carve((size_t)N * 8 * 4);
    float* sc1 = (float*)carve((size_t)E * 8 * 4);
    float* dn1 = (float*)carve((size_t)N * 8 * 4);
    float* hp1 = (float*)carve((size_t)N * 512 * 4);
    _Float16* Xh2 = (_Float16*)carve((size_t)N * 512 * 2);
    float* h2  = (float*)carve((size_t)N * 128 * 4);
    float* ss2 = (float*)carve((size_t)N * 4);
    float* st2 = (float*)carve((size_t)N * 4);
    float* sc2 = (float*)carve((size_t)E * 4);
    float* dn2 = (float*)carve((size_t)N * 4);
    float* hp2 = (float*)carve((size_t)N * 128 * 4);
    unsigned int* mk = (unsigned int*)carve(16 * 4);

    const int B = 256;
    auto grid = [](long long n, int b) { return (unsigned)((n + b - 1) / b); };

    // ---- layer 1 ----
    f32_to_f16<<<grid((long long)N * 128, B), B, 0, stream>>>(X, Xh, (long long)N * 128);
    pack_W1<<<grid(512 * 128, B), B, 0, stream>>>(W_heads, WT1);
    pack_W2<<<grid(128 * 512, B), B, 0, stream>>>(W_out, WT2);
    hipMemsetAsync(mk, 0, 16 * 4, stream);              // key(−NaN) = 0 == −inf
    hipMemsetAsync(dn1, 0, (size_t)N * 8 * 4, stream);
    hipMemsetAsync(hp1, 0, (size_t)N * 512 * 4, stream);
    hipMemsetAsync(dn2, 0, (size_t)N * 4, stream);
    hipMemsetAsync(hp2, 0, (size_t)N * 128 * 4, stream);

    {   // h1[N,512] = X @ Wcat
        long long waves = (long long)(N / 16) * (512 / 64);
        gemm_wmma<<<grid(waves * 32, B), B, 0, stream>>>(Xh, WT1, h1, N, 512, 128);
    }
    node_scores<<<grid((long long)N * 8, B), B, 0, stream>>>(h1, a_heads, ss1, st1, N, 8, 64, 512, 144);
    edge_scores<<<grid(E, B), B, 0, stream>>>(ei, eattr, a_heads, ss1, st1, sc1, mk, E, 8, 64, 144);
    exp_denom<<<grid((long long)E * 8, B), B, 0, stream>>>(sc1, mk, ei, dn1, E, 8);
    aggregate<<<grid((long long)E * 128, B), B, 0, stream>>>(ei, h1, sc1, dn1, hp1, E, 512, 64, 8);
    elu2_to_f16<<<grid((long long)N * 512, B), B, 0, stream>>>(hp1, Xh2, (long long)N * 512);

    // ---- layer 2 ----
    {   // h2[N,128] = x @ W_out
        long long waves = (long long)(N / 16) * (128 / 64);
        gemm_wmma<<<grid(waves * 32, B), B, 0, stream>>>(Xh2, WT2, h2, N, 128, 512);
    }
    node_scores<<<grid(N, B), B, 0, stream>>>(h2, a_out, ss2, st2, N, 1, 128, 128, 0);
    edge_scores<<<grid(E, B), B, 0, stream>>>(ei, eattr, a_out, ss2, st2, sc2, mk + 8, E, 1, 128, 0);
    exp_denom<<<grid(E, B), B, 0, stream>>>(sc2, mk + 8, ei, dn2, E, 1);
    aggregate<<<grid((long long)E * 32, B), B, 0, stream>>>(ei, h2, sc2, dn2, hp2, E, 128, 128, 1);
    elu_logsoftmax<<<grid((long long)N * 32, B), B, 0, stream>>>(hp2, out, N);
}
